// GraphPropLayer_22874995818875
// MI455X (gfx1250) — matrix operand order repaired and checked by reading, not compiled, not run on hardware
//
#include <hip/hip_runtime.h>
#include <hip/hip_bf16.h>

// ---------------------------------------------------------------------------
// GraphPropLayer on gfx1250: f16 WMMA (f32 accum) for all GEMMs,
// f32 global atomics for segment_sum, weights pre-packed in B-fragment order.
// Inner loops restructured so each B fragment feeds 4 independent WMMA chains
// (2 edge row-tiles x 2 directions), with all A fragments held in VGPRs.
// ---------------------------------------------------------------------------

typedef _Float16 half_t;
typedef half_t v8h  __attribute__((ext_vector_type(8)));
typedef half_t v16h __attribute__((ext_vector_type(16)));
typedef float  v8f  __attribute__((ext_vector_type(8)));

#define NNODES 50000
#define NEDGES 800000
#define DD 128
#define HH 256
#define ME 32            // edges per block

union HV16 { v16h v; v8h h[2]; };

// A-fragment (16x32 f16) from a row-major half array (rowStride in halves).
// Lane l: m = l&15, kh = (l>=16)*8 ; halves {kh..kh+7} and {kh+16..kh+23}.
__device__ __forceinline__ v16h load_a_frag(const half_t* base, int row0, int kbase,
                                            int rowStride, int lane) {
    int m  = lane & 15;
    int kh = (lane >> 4) << 3;
    const half_t* p = base + (size_t)(row0 + m) * rowStride + kbase + kh;
    HV16 u;
    u.h[0] = *(const v8h*)(p);
    u.h[1] = *(const v8h*)(p + 16);
    return u.v;
}

// B-fragment (32x16 f16) from packed weights: ((kt*ntiles+nt)*32+lane)*16 halves.
__device__ __forceinline__ v16h load_b_frag(const half_t* wp, int kt, int nt,
                                            int ntiles, int lane) {
    const half_t* p = wp + ((size_t)((kt * ntiles) + nt) * 32 + lane) * 16;
    HV16 u;
    u.h[0] = *(const v8h*)(p);
    u.h[1] = *(const v8h*)(p + 8);
    return u.v;
}

__device__ __forceinline__ v8f wmma_f16(v16h a, v16h b, v8f c) {
    return __builtin_amdgcn_wmma_f32_16x16x32_f16(false, a, false, b,
                                                  (short)0, c, false, false);
}

// ------------------------------- prep kernels ------------------------------

__global__ void zero_f32_kernel(float* __restrict__ p, int n) {
    int i = blockIdx.x * blockDim.x + threadIdx.x;
    if (i < n) p[i] = 0.0f;
}

__global__ void f32_to_f16_kernel(const float* __restrict__ in,
                                  half_t* __restrict__ out, int n) {
    int i = blockIdx.x * blockDim.x + threadIdx.x;
    if (i < n) out[i] = (half_t)in[i];
}

// Repack row-major f32 W[K][N] into WMMA B-fragment order (f16).
__global__ void pack_b_kernel(const float* __restrict__ W, int K, int N,
                              half_t* __restrict__ out) {
    int tid = blockIdx.x * blockDim.x + threadIdx.x;
    int ktiles = K >> 5, ntiles = N >> 4;
    int total = ktiles * ntiles * 32;
    if (tid >= total) return;
    int lane = tid & 31;
    int tile = tid >> 5;
    int tn = tile % ntiles;
    int tk = tile / ntiles;
    int khi = (lane >> 4) << 3;            // 0 or 8
    int n   = tn * 16 + (lane & 15);
    half_t* o = out + (size_t)tid * 16;
    #pragma unroll
    for (int j = 0; j < 16; ++j) {
        int k = tk * 32 + khi + (j & 7) + ((j >> 3) << 4);
        o[j] = (half_t)W[(size_t)k * N + n];
    }
}

// ------------------------------- edge kernel -------------------------------
// Block: 256 threads (8 waves), 32 edges. Wave w owns n-tiles {2w, 2w+1} and
// BOTH 16-edge row tiles; forward and reverse share the same A fragments
// (reverse K order is a rotation: AR[kt] == AF[(kt+4)&7]).
__global__ __launch_bounds__(256) void edge_kernel(
    const half_t* __restrict__ node_h,          // [NNODES][128] f16
    const float*  __restrict__ ef,              // [NEDGES]
    const int*    __restrict__ from_idx,
    const int*    __restrict__ to_idx,
    const half_t* __restrict__ W1p,  const half_t* __restrict__ W2p,
    const half_t* __restrict__ RW1p, const half_t* __restrict__ RW2p,
    const float*  __restrict__ W1full, const float* __restrict__ b1,
    const float*  __restrict__ b2,
    const float*  __restrict__ RW1full, const float* __restrict__ rb1,
    const float*  __restrict__ rb2,
    float* __restrict__ agg)                    // [NNODES][256] f32
{
    __shared__ __align__(16) half_t Afrom[ME][DD];
    __shared__ __align__(16) half_t Ato[ME][DD];
    __shared__ __align__(16) half_t Hf[ME][HH];
    __shared__ __align__(16) half_t Hr[ME][HH];
    __shared__ float efl[ME];
    __shared__ int   fid[ME];
    __shared__ int   tid2[ME];

    const int tid   = threadIdx.x;
    const int ebase = blockIdx.x * ME;

    if (tid < ME) {
        int e = ebase + tid;
        fid[tid]  = from_idx[e];
        tid2[tid] = to_idx[e];
        efl[tid]  = ef[e];
    }
    __syncthreads();

    // Gather node rows (16B chunks): 32 rows x 16 chunks, both matrices.
    for (int i = tid; i < ME * (DD / 8); i += 256) {
        int r = i >> 4;
        int c = (i & 15) << 3;
        *(v8h*)&Afrom[r][c] = *(const v8h*)&node_h[(size_t)fid[r]  * DD + c];
        *(v8h*)&Ato[r][c]   = *(const v8h*)&node_h[(size_t)tid2[r] * DD + c];
    }
    __syncthreads();

    const int wave = tid >> 5;
    const int lane = tid & 31;
    const int ntiles = HH / 16;

    const float* w1last  = W1full  + 256 * HH;   // row 256 (edge-feature row)
    const float* rw1last = RW1full + 256 * HH;

    // Preload all A fragments (forward K order) for both row tiles.
    v16h A0[8], A1[8];
    #pragma unroll
    for (int kt = 0; kt < 8; ++kt) {
        int kb = (kt & 3) << 5;
        const half_t* src = (kt < 4) ? &Afrom[0][0] : &Ato[0][0];
        A0[kt] = load_a_frag(src, 0,  kb, DD, lane);
        A1[kt] = load_a_frag(src, 16, kb, DD, lane);
    }

    // ---- Layer 1, both directions ----
    for (int t = 0; t < 2; ++t) {
        int nt = wave * 2 + t;
        v8f f0 = {}, f1 = {}, r0 = {}, r1 = {};
        #pragma unroll
        for (int kt = 0; kt < 8; ++kt) {
            v16h bf = load_b_frag(W1p,  kt, nt, ntiles, lane);
            v16h br = load_b_frag(RW1p, kt, nt, ntiles, lane);
            int kr = (kt + 4) & 7;
            f0 = wmma_f16(A0[kt], bf, f0);
            f1 = wmma_f16(A1[kt], bf, f1);
            r0 = wmma_f16(A0[kr], br, r0);
            r1 = wmma_f16(A1[kr], br, r1);
        }
        int n  = nt * 16 + (lane & 15);
        int mb = (lane >> 4) << 3;
        float b1n = b1[n],  w1n  = w1last[n];
        float rbn = rb1[n], rw1n = rw1last[n];
        #pragma unroll
        for (int r = 0; r < 8; ++r) {
            int m0 = mb + r, m1 = 16 + mb + r;
            float e0 = efl[m0], e1 = efl[m1];
            float vf0 = f0[r] + b1n + e0 * w1n;
            float vf1 = f1[r] + b1n + e1 * w1n;
            float vr0 = r0[r] + rbn + e0 * rw1n;
            float vr1 = r1[r] + rbn + e1 * rw1n;
            Hf[m0][n] = (half_t)(vf0 > 0.0f ? vf0 : 0.0f);
            Hf[m1][n] = (half_t)(vf1 > 0.0f ? vf1 : 0.0f);
            Hr[m0][n] = (half_t)(vr0 > 0.0f ? vr0 : 0.0f);
            Hr[m1][n] = (half_t)(vr1 > 0.0f ? vr1 : 0.0f);
        }
    }
    __syncthreads();

    // ---- Layer 2 + atomic segment-sum scatter ----
    for (int t = 0; t < 2; ++t) {
        int nt = wave * 2 + t;
        v8f f0 = {}, f1 = {}, r0 = {}, r1 = {};
        #pragma unroll
        for (int kt = 0; kt < 8; ++kt) {
            v16h bf = load_b_frag(W2p,  kt, nt, ntiles, lane);
            v16h br = load_b_frag(RW2p, kt, nt, ntiles, lane);
            int kb = kt << 5;
            f0 = wmma_f16(load_a_frag(&Hf[0][0], 0,  kb, HH, lane), bf, f0);
            f1 = wmma_f16(load_a_frag(&Hf[0][0], 16, kb, HH, lane), bf, f1);
            r0 = wmma_f16(load_a_frag(&Hr[0][0], 0,  kb, HH, lane), br, r0);
            r1 = wmma_f16(load_a_frag(&Hr[0][0], 16, kb, HH, lane), br, r1);
        }
        int n  = nt * 16 + (lane & 15);
        int mb = (lane >> 4) << 3;
        float b2n = b2[n], rbn = rb2[n];
        #pragma unroll
        for (int r = 0; r < 8; ++r) {
            int m0 = mb + r, m1 = 16 + mb + r;
            unsafeAtomicAdd(&agg[(size_t)tid2[m0] * HH + n], f0[r] + b2n);
            unsafeAtomicAdd(&agg[(size_t)tid2[m1] * HH + n], f1[r] + b2n);
            unsafeAtomicAdd(&agg[(size_t)fid[m0]  * HH + n], r0[r] + rbn);
            unsafeAtomicAdd(&agg[(size_t)fid[m1]  * HH + n], r1[r] + rbn);
        }
    }
}

// ------------------------------- node kernel -------------------------------
// Block: 256 threads (8 waves), 16 nodes. out = state + MLP([agg, state]).
// A fragments preloaded; 2 accumulator chains per B-load pair in layer 1.
__global__ __launch_bounds__(256) void node_kernel(
    const float*  __restrict__ agg,             // [NNODES][256] f32
    const half_t* __restrict__ node_h,          // [NNODES][128] f16
    const float*  __restrict__ node_f32,        // [NNODES][128] f32
    const half_t* __restrict__ Wn1p, const half_t* __restrict__ Wn2p,
    const float*  __restrict__ bn1,  const float* __restrict__ bn2,
    float* __restrict__ out)                    // [NNODES][128] f32
{
    __shared__ __align__(16) half_t Au[16][384];   // [agg | state], K = 384
    __shared__ __align__(16) half_t Hl[16][256];

    const int tid = threadIdx.x;
    const int nb  = blockIdx.x * 16;

    for (int i = tid; i < 16 * 256; i += 256) {
        int r = i >> 8, c = i & 255;
        Au[r][c] = (half_t)agg[(size_t)(nb + r) * HH + c];
    }
    for (int i = tid; i < 16 * 128; i += 256) {
        int r = i >> 7, c = i & 127;
        Au[r][256 + c] = node_h[(size_t)(nb + r) * DD + c];
    }
    __syncthreads();

    const int wave = tid >> 5;
    const int lane = tid & 31;

    // Layer 1: hidden[16][256]; wave owns ntiles {wave, wave+8}, two chains.
    {
        v16h Af[12];
        #pragma unroll
        for (int kt = 0; kt < 12; ++kt)
            Af[kt] = load_a_frag(&Au[0][0], 0, kt << 5, 384, lane);

        v8f a0 = {}, a1 = {};
        #pragma unroll
        for (int kt = 0; kt < 12; ++kt) {
            v16h b0 = load_b_frag(Wn1p, kt, wave,     16, lane);
            v16h b1v = load_b_frag(Wn1p, kt, wave + 8, 16, lane);
            a0 = wmma_f16(Af[kt], b0,  a0);
            a1 = wmma_f16(Af[kt], b1v, a1);
        }
        int n0 = wave * 16 + (lane & 15);
        int n1 = n0 + 128;
        int mb = (lane >> 4) << 3;
        float bn0 = bn1[n0], bn1v = bn1[n1];
        #pragma unroll
        for (int r = 0; r < 8; ++r) {
            float v0 = a0[r] + bn0;
            float v1 = a1[r] + bn1v;
            Hl[mb + r][n0] = (half_t)(v0 > 0.0f ? v0 : 0.0f);
            Hl[mb + r][n1] = (half_t)(v1 > 0.0f ? v1 : 0.0f);
        }
    }
    __syncthreads();

    // Layer 2: out[16][128]; each wave owns ntile = wave.
    {
        v16h Ah[8];
        #pragma unroll
        for (int kt = 0; kt < 8; ++kt)
            Ah[kt] = load_a_frag(&Hl[0][0], 0, kt << 5, 256, lane);

        v8f acc = {};
        #pragma unroll
        for (int kt = 0; kt < 8; ++kt)
            acc = wmma_f16(Ah[kt], load_b_frag(Wn2p, kt, wave, 8, lane), acc);

        int n  = wave * 16 + (lane & 15);
        int mb = (lane >> 4) << 3;
        float bn = bn2[n];
        #pragma unroll
        for (int r = 0; r < 8; ++r) {
            size_t m = (size_t)(nb + mb + r);
            out[m * DD + n] = node_f32[m * DD + n] + acc[r] + bn;
        }
    }
}

// ------------------------------- launcher ----------------------------------

extern "C" void kernel_launch(void* const* d_in, const int* in_sizes, int n_in,
                              void* d_out, int out_size, void* d_ws, size_t ws_size,
                              hipStream_t stream) {
    (void)in_sizes; (void)n_in; (void)out_size; (void)ws_size;

    const float* node_states   = (const float*)d_in[0];
    const float* edge_features = (const float*)d_in[1];
    const int*   from_idx      = (const int*)d_in[2];
    const int*   to_idx        = (const int*)d_in[3];
    const float* W1  = (const float*)d_in[4];
    const float* b1  = (const float*)d_in[5];
    const float* W2  = (const float*)d_in[6];
    const float* b2  = (const float*)d_in[7];
    const float* RW1 = (const float*)d_in[8];
    const float* Rb1 = (const float*)d_in[9];
    const float* RW2 = (const float*)d_in[10];
    const float* Rb2 = (const float*)d_in[11];
    const float* Wn1 = (const float*)d_in[12];
    const float* bn1 = (const float*)d_in[13];
    const float* Wn2 = (const float*)d_in[14];
    const float* bn2 = (const float*)d_in[15];
    float* out = (float*)d_out;

    char* ws = (char*)d_ws;
    size_t off = 0;
    auto walloc = [&](size_t bytes) -> void* {
        void* p = ws + off;
        off = (off + bytes + 255) & ~(size_t)255;
        return p;
    };

    float*  agg    = (float*)walloc((size_t)NNODES * HH * sizeof(float));
    half_t* node_h = (half_t*)walloc((size_t)NNODES * DD * sizeof(half_t));
    half_t* W1p    = (half_t*)walloc((size_t)256 * 256 * sizeof(half_t));
    half_t* W2p    = (half_t*)walloc((size_t)256 * 256 * sizeof(half_t));
    half_t* RW1p   = (half_t*)walloc((size_t)256 * 256 * sizeof(half_t));
    half_t* RW2p   = (half_t*)walloc((size_t)256 * 256 * sizeof(half_t));
    half_t* Wn1p   = (half_t*)walloc((size_t)384 * 256 * sizeof(half_t));
    half_t* Wn2p   = (half_t*)walloc((size_t)256 * 128 * sizeof(half_t));

    // 1) zero the aggregation buffer
    {
        int n = NNODES * HH;
        zero_f32_kernel<<<(n + 255) / 256, 256, 0, stream>>>(agg, n);
    }
    // 2) node states -> f16
    {
        int n = NNODES * DD;
        f32_to_f16_kernel<<<(n + 255) / 256, 256, 0, stream>>>(node_states, node_h, n);
    }
    // 3) repack weights to B-fragment order (matrix parts only; W1/RW1 row 256
    //    is the edge-feature rank-1 term handled in the VALU epilogue)
    {
        auto pack = [&](const float* W, int K, int N, half_t* o) {
            int total = (K >> 5) * (N >> 4) * 32;
            pack_b_kernel<<<(total + 255) / 256, 256, 0, stream>>>(W, K, N, o);
        };
        pack(W1,  256, 256, W1p);
        pack(W2,  256, 256, W2p);
        pack(RW1, 256, 256, RW1p);
        pack(RW2, 256, 256, RW2p);
        pack(Wn1, 384, 256, Wn1p);
        pack(Wn2, 256, 128, Wn2p);
    }
    // 4) edge message passing (both directions) + atomic segment sums
    edge_kernel<<<NEDGES / ME, 256, 0, stream>>>(
        node_h, edge_features, from_idx, to_idx,
        W1p, W2p, RW1p, RW2p,
        W1, b1, b2, RW1, Rb1, Rb2, agg);
    // 5) node update MLP with residual
    node_kernel<<<NNODES / 16, 256, 0, stream>>>(
        agg, node_h, node_states, Wn1p, Wn2p, bn1, bn2, out);
}
